// DINOv3ImageLevelDetector_66554813219120
// MI455X (gfx1250) — matrix operand order repaired and checked by reading, not compiled, not run on hardware
//
#include <hip/hip_runtime.h>
#include <hip/hip_bf16.h>

typedef _Float16 v16h __attribute__((ext_vector_type(16)));
typedef _Float16 v8h  __attribute__((ext_vector_type(8)));
typedef float    v8f  __attribute__((ext_vector_type(8)));

#define B_ROWS      1024
#define N_COLS      50000
#define DIM         768
#define TOPK        9
#define NTILES_N    3125        // N_COLS / 16 (exact)
#define CHUNK_TILES 32
#define CHUNK_COLS  512
#define NCHUNKS     98          // ceil(3125 / 32)
#define LDS_STRIDE  516         // 8*516 % 64 == 32 -> half-waves hit disjoint banks
#define FLT_BIG     3.4028235e38f

#define WMMA_F16(a, b, c) __builtin_amdgcn_wmma_f32_16x16x32_f16( \
        false, (a), false, (b), (short)0, (c), false, false)

__device__ __forceinline__ float wave_min32(float v) {
#pragma unroll
    for (int off = 16; off >= 1; off >>= 1)
        v = fminf(v, __shfl_xor(v, off, 32));
    return v;
}

// fp32 -> f16 copy + per-row squared L2 norm. blockDim = 256 (8 waves).
__global__ void prep_f16_sq(const float* __restrict__ src,
                            _Float16* __restrict__ dst,
                            float* __restrict__ sq) {
    const int row = blockIdx.x;
    const float* s = src + (size_t)row * DIM;
    _Float16*    d = dst + (size_t)row * DIM;
    float acc = 0.f;
    for (int k = threadIdx.x; k < DIM; k += 256) {
        float v = s[k];
        d[k] = (_Float16)v;
        acc = fmaf(v, v, acc);
    }
#pragma unroll
    for (int off = 16; off >= 1; off >>= 1) acc += __shfl_xor(acc, off, 32);
    __shared__ float r[8];
    if ((threadIdx.x & 31) == 0) r[threadIdx.x >> 5] = acc;
    __syncthreads();
    if (threadIdx.x == 0) {
        float t = 0.f;
#pragma unroll
        for (int i = 0; i < 8; ++i) t += r[i];
        sq[row] = t;
    }
}

// One block = 16 rows x 512 cols of the distance matrix. 4 waves.
// Each wave: 8 contiguous col-tiles, processed as 2 groups of 4 with
// 4 accumulators so each A fragment is reused across 4 WMMAs.
__global__ void __launch_bounds__(128)
gemm_dist_topk(const _Float16* __restrict__ Ah,
               const _Float16* __restrict__ Mh,
               const float* __restrict__ Xsq,
               const float* __restrict__ Msq,
               float* __restrict__ partial) {
    __shared__ float sdist[16 * LDS_STRIDE];

    const int lane    = threadIdx.x & 31;
    const int wid     = threadIdx.x >> 5;
    const int rowBase = blockIdx.x * 16;     // 0..1008
    const int chunk   = blockIdx.y;          // 0..97

    const int nlane = lane & 15;
    const int kA    = (lane >> 4) << 4;      // A frag: +16 halfs on upper half-wave
    const int kB    = (lane >> 4) << 4;      // B frag: +16 halfs on upper half-wave
    const int half  = (lane >> 4) << 3;      // C/D: M offset for upper half-wave

    // Preload x^2 for this lane's 8 output rows (constant across tiles).
    float xq[8];
#pragma unroll
    for (int v = 0; v < 8; ++v) xq[v] = Xsq[rowBase + v + half];

    const _Float16* arow = Ah + (size_t)(rowBase + nlane) * DIM;
    const int tBase = chunk * CHUNK_TILES + wid * 8;   // 8 tiles per wave

    for (int g = 0; g < 2; ++g) {
        const int t0 = tBase + g * 4;
        bool valid[4];
        int  nG[4];
        const _Float16* brow[4];
#pragma unroll
        for (int j = 0; j < 4; ++j) {
            const int tg = t0 + j;                     // wave-uniform
            valid[j] = (tg < NTILES_N);
            nG[j]    = (valid[j] ? tg : 0) * 16 + nlane;
            brow[j]  = Mh + (size_t)nG[j] * DIM;
        }

        v8f acc0 = {}, acc1 = {}, acc2 = {}, acc3 = {};
#pragma unroll 2
        for (int k0 = 0; k0 < DIM; k0 += 32) {
            v8h a0 = *(const v8h*)(arow + k0 + kA);
            v8h a1 = *(const v8h*)(arow + k0 + kA + 8);
            v16h a = __builtin_shufflevector(a0, a1,
                     0,1,2,3,4,5,6,7,8,9,10,11,12,13,14,15);
            v16h b0 = *(const v16h*)(brow[0] + k0 + kB);
            v16h b1 = *(const v16h*)(brow[1] + k0 + kB);
            v16h b2 = *(const v16h*)(brow[2] + k0 + kB);
            v16h b3 = *(const v16h*)(brow[3] + k0 + kB);
            acc0 = WMMA_F16(a, b0, acc0);
            acc1 = WMMA_F16(a, b1, acc1);
            acc2 = WMMA_F16(a, b2, acc2);
            acc3 = WMMA_F16(a, b3, acc3);
        }

        auto epilogue = [&](int j, const v8f& acc) {
            const int tLocal = wid * 8 + g * 4 + j;    // 0..31 within chunk
            if (valid[j]) {
                const float mq = Msq[nG[j]];
#pragma unroll
                for (int v = 0; v < 8; ++v) {
                    float d2   = xq[v] + mq - 2.0f * acc[v];
                    float dist = sqrtf(fmaxf(d2, 0.0f));
                    sdist[(v + half) * LDS_STRIDE + tLocal * 16 + nlane] = dist;
                }
            } else {
#pragma unroll
                for (int v = 0; v < 8; ++v)
                    sdist[(v + half) * LDS_STRIDE + tLocal * 16 + nlane] = FLT_BIG;
            }
        };
        epilogue(0, acc0);
        epilogue(1, acc1);
        epilogue(2, acc2);
        epilogue(3, acc3);
    }
    __syncthreads();

    // Selection: wave `wid` handles rows 4*wid .. 4*wid+3.
    for (int rr = 0; rr < 4; ++rr) {
        const int row = wid * 4 + rr;
        float t[9];
#pragma unroll
        for (int j = 0; j < 9; ++j) t[j] = FLT_BIG;
        for (int c = lane; c < CHUNK_COLS; c += 32) {
            float val = sdist[row * LDS_STRIDE + c];
            if (val < t[8]) {            // branch-free sorted insert
                float x = val;
#pragma unroll
                for (int j = 0; j < 9; ++j) {
                    float lo = fminf(t[j], x);
                    float hi = fmaxf(t[j], x);
                    t[j] = lo; x = hi;
                }
            }
        }
        // 9-round cross-lane pop-merge -> top-9 of the chunk.
        float* outp = partial + ((size_t)(rowBase + row) * NCHUNKS + chunk) * TOPK;
#pragma unroll
        for (int j = 0; j < TOPK; ++j) {
            float cand = t[0];
            float mn   = wave_min32(cand);
            unsigned long long bal = __ballot(cand == mn);
            int owner = __ffsll(bal) - 1;
            if (lane == owner) {
#pragma unroll
                for (int q = 0; q < 8; ++q) t[q] = t[q + 1];
                t[8] = FLT_BIG;
            }
            if (lane == 0) outp[j] = mn;
        }
    }
}

// One wave per row: merge 98 partial top-9 lists -> top-9 -> mean.
__global__ void __launch_bounds__(128)
final_topk_mean(const float* __restrict__ partial, float* __restrict__ out) {
    const int lane = threadIdx.x & 31;
    const int wid  = threadIdx.x >> 5;
    const int row  = blockIdx.x * 4 + wid;          // grid 256 * 4 waves = 1024
    const float* p = partial + (size_t)row * (NCHUNKS * TOPK);

    float t[9];
#pragma unroll
    for (int j = 0; j < 9; ++j) t[j] = FLT_BIG;
    for (int c = lane; c < NCHUNKS * TOPK; c += 32) {
        float val = p[c];
        if (val < t[8]) {
            float x = val;
#pragma unroll
            for (int j = 0; j < 9; ++j) {
                float lo = fminf(t[j], x);
                float hi = fmaxf(t[j], x);
                t[j] = lo; x = hi;
            }
        }
    }
    float sum = 0.f;
#pragma unroll
    for (int j = 0; j < TOPK; ++j) {
        float cand = t[0];
        float mn   = wave_min32(cand);
        unsigned long long bal = __ballot(cand == mn);
        int owner = __ffsll(bal) - 1;
        if (lane == owner) {
#pragma unroll
            for (int q = 0; q < 8; ++q) t[q] = t[q + 1];
            t[8] = FLT_BIG;
        }
        sum += mn;
    }
    if (lane == 0) out[row] = sum * (1.0f / (float)TOPK);
}

extern "C" void kernel_launch(void* const* d_in, const int* in_sizes, int n_in,
                              void* d_out, int out_size, void* d_ws, size_t ws_size,
                              hipStream_t stream) {
    const float* features = (const float*)d_in[0];   // [1024, 768] fp32
    const float* memory   = (const float*)d_in[1];   // [50000, 768] fp32
    // d_in[2] == k == 9 (hardcoded)
    float* out = (float*)d_out;                      // [1024] fp32

    char* ws = (char*)d_ws;
    size_t off = 0;
    _Float16* Ah  = (_Float16*)(ws + off); off += (size_t)B_ROWS * DIM * 2;           // 1.5 MB
    _Float16* Mh  = (_Float16*)(ws + off); off += (size_t)N_COLS * DIM * 2;           // 76.8 MB
    float*    XsqP= (float*)(ws + off);    off += (size_t)B_ROWS * 4;                 // 4 KB
    float*    MsqP= (float*)(ws + off);    off += (((size_t)N_COLS * 4) + 255) & ~(size_t)255;
    float*    part= (float*)(ws + off);    // 1024*98*9*4 = 3.6 MB

    prep_f16_sq<<<B_ROWS, 256, 0, stream>>>(features, Ah, XsqP);
    prep_f16_sq<<<N_COLS, 256, 0, stream>>>(memory,   Mh, MsqP);
    gemm_dist_topk<<<dim3(B_ROWS / 16, NCHUNKS), 128, 0, stream>>>(Ah, Mh, XsqP, MsqP, part);
    final_topk_mean<<<B_ROWS / 4, 128, 0, stream>>>(part, out);
}
